// LocalRNN_12068858101892
// MI455X (gfx1250) — matrix operand order repaired and checked by reading, not compiled, not run on hardware
//
#include <hip/hip_runtime.h>

typedef __attribute__((ext_vector_type(16))) __bf16 v16bf;
typedef __attribute__((ext_vector_type(8)))  __bf16 v8bf;
typedef __attribute__((ext_vector_type(4)))  __bf16 v4bf;
typedef __attribute__((ext_vector_type(8)))  float  v8f;
typedef __attribute__((ext_vector_type(4)))  float  v4f;
typedef __attribute__((ext_vector_type(4)))  unsigned int u32x4;
typedef __attribute__((ext_vector_type(8)))  int    i32x8;
typedef __attribute__((ext_vector_type(4)))  int    i32x4;

#define Bsz 16
#define Lsz 2048
#define Dsz 128
#define KS  8
#define NW  49152   // 384*128 elements per weight matrix

union FragU { v16bf v; v8bf h[2]; };

// A fragment (16x32 bf16): lane holds row M=l15, K = kc*32 + (hi?8:0) + {0..7,16..23}
__device__ __forceinline__ v16bf a_frag_bf16(const __bf16* row, int kc, int half) {
  const int k0 = kc * 32 + half * 8;
  FragU f;
  f.h[0] = *(const v8bf*)(row + k0);
  f.h[1] = *(const v8bf*)(row + k0 + 16);
  return f.v;
}

// B fragment (32x16 bf16): lane holds column N=l15, K = kc*32 + (hi?8:0) + {0..7,16..23}
__device__ __forceinline__ v16bf b_frag(const __bf16* row, int kc, int half) {
  const int k0 = kc * 32 + half * 8;
  FragU f;
  f.h[0] = *(const v8bf*)(row + k0);
  f.h[1] = *(const v8bf*)(row + k0 + 16);
  return f.v;
}

__device__ __forceinline__ v8f wmma_bf16(v16bf a, v16bf b, v8f c) {
  return __builtin_amdgcn_wmma_f32_16x16x32_bf16(false, a, false, b, (short)0, c, false, false);
}

__device__ __forceinline__ float sigm(float x)      { return 1.0f / (1.0f + __expf(-x)); }
__device__ __forceinline__ float tanh_fast(float x) { return 2.0f / (1.0f + __expf(-2.0f * x)) - 1.0f; }

// ---- Prep: convert W_hh and W_ih (f32) to bf16 in workspace, once ----
__global__ __launch_bounds__(256) void convert_weights_kernel(
    const float* __restrict__ Whh, const float* __restrict__ Wih,
    __bf16* __restrict__ wbf) {
  const int i = (blockIdx.x * 256 + threadIdx.x) * 4;   // 2*NW elements total
  const float* src = (i < NW) ? (Whh + i) : (Wih + (i - NW));
  v4f t = *(const v4f*)src;
  v4bf o;
  o[0] = (__bf16)t[0]; o[1] = (__bf16)t[1]; o[2] = (__bf16)t[2]; o[3] = (__bf16)t[3];
  *(v4bf*)(wbf + i) = o;
}

__global__ __launch_bounds__(256) void localrnn_gru_kernel(
    const float*  __restrict__ x,
    const __bf16* __restrict__ Wih_bf,   // [384][128] bf16 (in ws)
    const __bf16* __restrict__ Whh_bf,   // [384][128] bf16 (in ws)
    const float*  __restrict__ bih, const float* __restrict__ bhh,
    float* __restrict__ out)
{
  __shared__ __align__(16) float  sX[32 * Dsz];        // 16 KB: x rows t0-16..t0+15 (TDM)
  __shared__ __align__(16) __bf16 sXbf[32 * Dsz];      //  8 KB: same tile, bf16 (one pass)
  __shared__ __align__(16) float  sG4[24 * Dsz * 4];   // 48 KB: gates packed {r,z,n,pad}
  __shared__ __align__(16) __bf16 sH[16 * Dsz];        //  4 KB: hidden state (bf16 A frags)

  const int tid  = threadIdx.x;
  const int w    = tid >> 5;
  const int lane = tid & 31;
  const int half = lane >> 4;
  const int l15  = lane & 15;

  const int r0 = blockIdx.x * 16;
  const int bi = r0 / Lsz;
  const int t0 = r0 % Lsz;

  // rows t0-16 .. t0+15; clamp the front pad (t<0) out of the DMA
  const int tlo    = t0 - 16;
  const int tstart = tlo < 0 ? 0 : tlo;
  const int pad    = tstart - tlo;
  const int nrows  = 32 - pad;

  // ---- Phase 0a: TDM DMA of the contiguous x tile into LDS (wave 0 issues) ----
  if (w == 0) {
    const unsigned long long ga =
        (unsigned long long)(uintptr_t)(x + ((size_t)bi * Lsz + tstart) * Dsz);
    const unsigned ldsa = (unsigned)(uintptr_t)(&sX[pad * Dsz]);
    const unsigned dim0 = (unsigned)(nrows * Dsz);       // elements (data_size = 4B)
    u32x4 g0;
    g0[0] = 1u;                                          // count=1, user descriptor
    g0[1] = ldsa;                                        // lds_addr
    g0[2] = (unsigned)(ga & 0xFFFFFFFFull);              // global_addr[31:0]
    g0[3] = (unsigned)((ga >> 32) & 0x01FFFFFFull) | (2u << 30);  // addr[56:32] | type=2
    i32x8 g1;
    g1[0] = (int)(2u << 16);                             // data_size=2 -> 4 bytes
    g1[1] = (int)((dim0 & 0xFFFFu) << 16);               // tensor_dim0[15:0]
    g1[2] = (int)((dim0 >> 16) | (1u << 16));            // tensor_dim0[31:16] | tensor_dim1=1
    g1[3] = (int)(dim0 << 16);                           // tile_dim0 (1-D tile)
    g1[4] = 0;
    g1[5] = (int)dim0;                                   // tensor_dim0_stride[31:0]
    g1[6] = 0;
    g1[7] = 0;
    i32x4 gz4 = {0, 0, 0, 0};
    i32x8 gz8 = {0, 0, 0, 0, 0, 0, 0, 0};
    __builtin_amdgcn_tensor_load_to_lds(g0, g1, gz4, gz4, gz8, 0);
  }

  // ---- Phase 0b: zero hidden state and the padded x rows ----
  for (int i = tid; i < 16 * Dsz; i += 256) sH[i] = (__bf16)0.0f;
  for (int i = tid; i < pad * Dsz; i += 256) sX[i] = 0.0f;

  if (w == 0) __builtin_amdgcn_s_wait_tensorcnt(0);
  __syncthreads();

  // ---- Phase 0c: one cooperative f32->bf16 pass over the x tile ----
  for (int i = tid * 8; i < 32 * Dsz; i += 256 * 8) {
    v4f t0v = *(const v4f*)(sX + i);
    v4f t1v = *(const v4f*)(sX + i + 4);
    v8bf o;
    o[0] = (__bf16)t0v[0]; o[1] = (__bf16)t0v[1]; o[2] = (__bf16)t0v[2]; o[3] = (__bf16)t0v[3];
    o[4] = (__bf16)t1v[0]; o[5] = (__bf16)t1v[1]; o[6] = (__bf16)t1v[2]; o[7] = (__bf16)t1v[3];
    *(v8bf*)(sXbf + i) = o;
  }
  __syncthreads();

  // Wave w owns gate columns 16w..16w+16 of each of r/z/n
  const int col = w * 16 + l15;

  // ---- Phase 1: G tile = x_window @ W_ih^T + b_ih (+ b_hh for r,z folded in) ----
  {
    v8f c[2][3];
    const float bR = bih[col] + bhh[col];                // r: b_ih + b_hh (linear)
    const float bZ = bih[128 + col] + bhh[128 + col];    // z: b_ih + b_hh (linear)
    const float bN = bih[256 + col];                     // n: only b_ih (b_hh_n is gated by r)
    #pragma unroll
    for (int mt = 0; mt < 2; mt++) {
      #pragma unroll
      for (int j = 0; j < 8; j++) { c[mt][0][j] = bR; c[mt][1][j] = bZ; c[mt][2][j] = bN; }
    }
    const __bf16* xr0 = sXbf + l15 * Dsz;
    const __bf16* xr1 = sXbf + (16 + l15) * Dsz;
    const __bf16* wr  = Wih_bf + (size_t)(w * 16 + l15) * Dsz;
    const __bf16* wz  = Wih_bf + (size_t)((w + 8) * 16 + l15) * Dsz;
    const __bf16* wn  = Wih_bf + (size_t)((w + 16) * 16 + l15) * Dsz;
    #pragma unroll
    for (int kc = 0; kc < 4; kc++) {
      v16bf a0 = a_frag_bf16(xr0, kc, half);
      v16bf a1 = a_frag_bf16(xr1, kc, half);
      v16bf fR = b_frag(wr, kc, half);
      v16bf fZ = b_frag(wz, kc, half);
      v16bf fN = b_frag(wn, kc, half);
      c[0][0] = wmma_bf16(a0, fR, c[0][0]);
      c[0][1] = wmma_bf16(a0, fZ, c[0][1]);
      c[0][2] = wmma_bf16(a0, fN, c[0][2]);
      c[1][0] = wmma_bf16(a1, fR, c[1][0]);
      c[1][1] = wmma_bf16(a1, fZ, c[1][1]);
      c[1][2] = wmma_bf16(a1, fN, c[1][2]);
    }
    // gate-interleaved store: one b128 per row
    #pragma unroll
    for (int mt = 0; mt < 2; mt++) {
      #pragma unroll
      for (int j = 0; j < 8; j++) {
        const int m  = mt * 16 + j + half * 8;
        const int sr = m - 9;                       // sG row 0 <-> t0-7
        if (sr >= 0 && sr < 24) {
          v4f g = { c[mt][0][j], c[mt][1][j], c[mt][2][j], 0.0f };
          *(v4f*)(sG4 + (sr * Dsz + col) * 4) = g;
        }
      }
    }
  }

  // ---- Hoist W_hh B-fragments into registers (invariant across all 8 steps) ----
  v16bf rbR[4], rbZ[4], rbN[4];
  {
    const __bf16* hr = Whh_bf + (size_t)(w * 16 + l15) * Dsz;
    const __bf16* hz = Whh_bf + (size_t)((w + 8) * 16 + l15) * Dsz;
    const __bf16* hn = Whh_bf + (size_t)((w + 16) * 16 + l15) * Dsz;
    #pragma unroll
    for (int kc = 0; kc < 4; kc++) {
      rbR[kc] = b_frag(hr, kc, half);
      rbZ[kc] = b_frag(hz, kc, half);
      rbN[kc] = b_frag(hn, kc, half);
    }
  }
  __syncthreads();   // sG4 (and sH zero) visible to all waves

  // ---- Phase 2: 8-step GRU recurrence ----
  const float bhhN = bhh[256 + col];
  v8f bhhN8;
  #pragma unroll
  for (int j = 0; j < 8; j++) bhhN8[j] = bhhN;

  v8f h;
  #pragma unroll
  for (int j = 0; j < 8; j++) h[j] = 0.0f;

  const __bf16* aRow = sH + l15 * Dsz;

  for (int k = 0; k < KS; k++) {
    v8f aR = {}, aZ = {};      // zero C -> first WMMA can use inline-0 SRC2
    v8f aN = bhhN8;            // b_hh_n must ride inside r*(...) term
    #pragma unroll
    for (int kc = 0; kc < 4; kc++) {
      v16bf a = a_frag_bf16(aRow, kc, half);
      aR = wmma_bf16(a, rbR[kc], aR);
      aZ = wmma_bf16(a, rbZ[kc], aZ);
      aN = wmma_bf16(a, rbN[kc], aN);
    }
    __syncthreads();   // all reads of sH done before overwrite

    #pragma unroll
    for (int j = 0; j < 8; j++) {
      const int m = j + half * 8;
      v4f g = *(const v4f*)(sG4 + ((m + k) * Dsz + col) * 4);
      const float r  = sigm(g[0] + aR[j]);
      const float z  = sigm(g[1] + aZ[j]);
      const float n  = tanh_fast(g[2] + r * aN[j]);
      const float hv = (1.0f - z) * n + z * h[j];
      h[j] = hv;
      sH[m * Dsz + col] = (__bf16)hv;
    }
    __syncthreads();   // new h visible before next step
  }

  // ---- Final hidden state -> output (f32) ----
  #pragma unroll
  for (int j = 0; j < 8; j++) {
    const int m = j + half * 8;
    out[((size_t)(r0 + m)) * Dsz + col] = h[j];
  }
}

extern "C" void kernel_launch(void* const* d_in, const int* in_sizes, int n_in,
                              void* d_out, int out_size, void* d_ws, size_t ws_size,
                              hipStream_t stream) {
  (void)in_sizes; (void)n_in; (void)out_size; (void)ws_size;
  const float* x   = (const float*)d_in[0];
  const float* Wih = (const float*)d_in[1];
  const float* Whh = (const float*)d_in[2];
  const float* bih = (const float*)d_in[3];
  const float* bhh = (const float*)d_in[4];
  float* out = (float*)d_out;

  __bf16* wbf = (__bf16*)d_ws;           // [0,NW): W_hh bf16, [NW,2*NW): W_ih bf16
  __bf16* Whh_bf = wbf;
  __bf16* Wih_bf = wbf + NW;

  hipLaunchKernelGGL(convert_weights_kernel, dim3((2 * NW) / (256 * 4)), dim3(256), 0, stream,
                     Whh, Wih, wbf);
  hipLaunchKernelGGL(localrnn_gru_kernel, dim3((Bsz * Lsz) / 16), dim3(256), 0, stream,
                     x, Wih_bf, Whh_bf, bih, bhh, out);
}
